// GRUClassifier_2259152798211
// MI455X (gfx1250) — matrix operand order repaired and checked by reading, not compiled, not run on hardware
//
#include <hip/hip_runtime.h>
#include <hip/hip_bf16.h>
#include <math.h>

// GRU classifier for MI455X (gfx1250), wave32 / WMMA.
// T=512, B=512, H=128. Batch rows are independent through the recurrence,
// so each block owns a 16-row batch tile + one direction and scans all 512
// steps with W_hh resident in VGPRs and h resident in LDS.

#define TT   512
#define BB   512
#define HH   128
#define G3   384     // 3*H gate columns
#define BT   16      // batch-tile rows per block (WMMA M)
#define NCOL 48      // gate columns per wave (3 N-tiles of 16)

typedef __attribute__((ext_vector_type(16))) __bf16 v16bf;
typedef __attribute__((ext_vector_type(8)))  __bf16 v8bf;
typedef __attribute__((ext_vector_type(8)))  float  v8f;

// ---- CDNA5 async global->LDS (guarded; falls back to direct loads) --------
#if defined(__HIP_DEVICE_COMPILE__) && __has_builtin(__builtin_amdgcn_global_load_async_to_lds_b128)
#define HAVE_ASYNC 1
#else
#define HAVE_ASYNC 0
#endif

#if HAVE_ASYNC
#if __has_builtin(__builtin_amdgcn_s_wait_asynccnt)
#define WAIT_ASYNC(n) __builtin_amdgcn_s_wait_asynccnt(n)
#else
#define WAIT_ASYNC(n) asm volatile("s_wait_asynccnt %0" :: "n"(n) : "memory")
#endif
// builtin takes typed int4 pointers: (AS1 v4i* src, AS3 v4i* dst, imm, imm)
typedef __attribute__((__vector_size__(16))) int v4i_t;
typedef __attribute__((address_space(1))) v4i_t* g_v4i_p;
typedef __attribute__((address_space(3))) v4i_t* l_v4i_p;
// one wave32 instruction moves 512B (16B per lane): exactly one y0 row
static __device__ inline void async_row_copy16(const __bf16* gsrc, __bf16* ldst, int lane) {
    __builtin_amdgcn_global_load_async_to_lds_b128(
        (g_v4i_p)(gsrc + lane * 8),
        (l_v4i_p)(ldst + lane * 8), 0, 0);
}
#endif

static __device__ inline v8f vzero8() {
    v8f r;
    for (int i = 0; i < 8; ++i) r[i] = 0.f;
    return r;
}
static __device__ inline v16bf joinbf(v8bf lo, v8bf hi) {
    return __builtin_shufflevector(lo, hi, 0,1,2,3,4,5,6,7,8,9,10,11,12,13,14,15);
}
static __device__ inline v8f wmma_bf16(v16bf a, v16bf b, v8f c) {
    // D(16x16,f32) = A(16x32,bf16) * B(32x16,bf16) + C
    return __builtin_amdgcn_wmma_f32_16x16x32_bf16(false, a, false, b, (short)0, c, false, false);
}
static __device__ inline float sigm(float x)      { return 1.f / (1.f + __expf(-x)); }
static __device__ inline float tanh_fast(float x) { float e = __expf(-2.f * x); return (1.f - e) / (1.f + e); }

// ---------------------------------------------------------------------------
// Layer-0 scan: input dim 4 (x projection done in VALU during the gate phase),
// recurrent K=128 via WMMA with W_hh in registers. Writes y0[t][b][dir*128+j]
// as bf16 to workspace.
// ---------------------------------------------------------------------------
extern "C" __global__ __launch_bounds__(256)
void gru_l0_scan(const float* __restrict__ x,
                 const float* __restrict__ wih_f, const float* __restrict__ whh_f,
                 const float* __restrict__ bih_f, const float* __restrict__ bhh_f,
                 const float* __restrict__ wih_b, const float* __restrict__ whh_b,
                 const float* __restrict__ bih_b, const float* __restrict__ bhh_b,
                 __bf16* __restrict__ y0)
{
    extern __shared__ char smem[];
    float*  wihL = (float*)smem;           // [384][4]
    float*  bihL = wihL + G3 * 4;          // [384]
    float*  bhhL = bihL + G3;              // [384]
    float*  xtL  = bhhL + G3;              // [16][4]
    float*  gbuf = xtL + BT * 4;           // [16][384] hidden-proj spill
    float*  hF   = gbuf + BT * G3;         // [16][128] h (f32)
    __bf16* hB   = (__bf16*)(hF + BT * HH);// [16][128] h (bf16, WMMA A source)

    const int tid   = threadIdx.x;
    const int lane  = tid & 31;
    const int wave  = tid >> 5;
    const int dir   = blockIdx.x & 1;
    const int bbase = (blockIdx.x >> 1) * BT;

    const float* wih = dir ? wih_b : wih_f;
    const float* whh = dir ? whh_b : whh_f;
    const float* bih = dir ? bih_b : bih_f;
    const float* bhh = dir ? bhh_b : bhh_f;

    for (int i = tid; i < G3 * 4; i += 256) wihL[i] = wih[i];
    for (int i = tid; i < G3; i += 256) { bihL[i] = bih[i]; bhhL[i] = bhh[i]; }
    for (int i = tid; i < BT * HH; i += 256) { hF[i] = 0.f; hB[i] = (__bf16)0.f; }

    const int n_lane = lane & 15;          // N (and A-row M) within a 16-tile
    const int kB = (lane >> 4) << 4;       // B operand: K half 0 / 16
    const int kA = (lane >> 4) << 4;       // A operand: K half 0 / 16

    // W_hh slice resident in VGPRs for the whole scan: 3 N-tiles x 4 K-steps.
    // B layout: lane L holds column N=L&15, K = kB .. kB+15 (contiguous).
    v16bf Bw[3][4];
#pragma unroll
    for (int i = 0; i < 3; ++i) {
        const int N = wave * NCOL + i * 16 + n_lane;
#pragma unroll
        for (int kk = 0; kk < 4; ++kk) {
            const float* src = whh + (size_t)N * HH + kk * 32 + kB;
            v16bf t;
#pragma unroll
            for (int e = 0; e < 16; ++e) t[e] = (__bf16)src[e];
            Bw[i][kk] = t;
        }
    }
    __syncthreads();

    for (int s = 0; s < TT; ++s) {
        const int tstep = dir ? (TT - 1 - s) : s;

        // stage x_t tile [16][4]
        if (tid < BT * 4) {
            const int m = tid >> 2, k = tid & 3;
            xtL[tid] = x[((size_t)(bbase + m) * TT + tstep) * 4 + k];
        }
        __syncthreads();

        // A operands from h (bf16): lane holds row M=n_lane, K = kA .. kA+15 (contiguous)
        v16bf A[4];
#pragma unroll
        for (int kk = 0; kk < 4; ++kk) {
            const __bf16* hr = hB + n_lane * HH + kk * 32 + kA;
            A[kk] = joinbf(*(const v8bf*)hr, *(const v8bf*)(hr + 8));
        }
        v8f acc[3];
#pragma unroll
        for (int i = 0; i < 3; ++i) acc[i] = vzero8();
#pragma unroll
        for (int kk = 0; kk < 4; ++kk)
#pragma unroll
            for (int i = 0; i < 3; ++i)
                acc[i] = wmma_bf16(A[kk], Bw[i][kk], acc[i]);

        // spill C: VGPR v -> row (v + 8*(lane>=16)), col N
        const int mrow = (lane >> 4) << 3;
#pragma unroll
        for (int i = 0; i < 3; ++i) {
            const int N = wave * NCOL + i * 16 + n_lane;
#pragma unroll
            for (int v = 0; v < 8; ++v) gbuf[(mrow + v) * G3 + N] = acc[i][v];
        }
        __syncthreads();

        // gates (r,z,n) + state update; x projection is a dot-4 in VALU
        for (int e = tid; e < BT * HH; e += 256) {
            const int m = e >> 7, j = e & 127;
            const float hr_ = gbuf[m * G3 + j      ] + bhhL[j      ];
            const float hz_ = gbuf[m * G3 + 128 + j] + bhhL[128 + j];
            const float hn_ = gbuf[m * G3 + 256 + j] + bhhL[256 + j];
            float xr = bihL[j], xz = bihL[128 + j], xn = bihL[256 + j];
#pragma unroll
            for (int k = 0; k < 4; ++k) {
                const float xv = xtL[m * 4 + k];
                xr += xv * wihL[(j      ) * 4 + k];
                xz += xv * wihL[(128 + j) * 4 + k];
                xn += xv * wihL[(256 + j) * 4 + k];
            }
            const float r = sigm(xr + hr_);
            const float z = sigm(xz + hz_);
            const float n = tanh_fast(xn + r * hn_);
            const float hnew = (1.f - z) * n + z * hF[e];
            hF[e] = hnew;
            const __bf16 hb = (__bf16)hnew;
            hB[e] = hb;
            y0[((size_t)tstep * BB + (bbase + m)) * 256 + dir * HH + j] = hb;
        }
        __syncthreads();
    }
}

// ---------------------------------------------------------------------------
// Layer-1 scan: input dim 256. The input projection (K=256) is fused into the
// scan via WMMA with W_ih_l1 resident in LDS (192 KB bf16 — needs the CDNA5
// 320KB WGP LDS). Hidden projection (K=128) keeps W_hh_l1 in registers.
// y0[t] rows are double-buffered into LDS with async global->LDS loads,
// overlapped with the hidden-projection WMMAs. Only final h is written out.
// ---------------------------------------------------------------------------
extern "C" __global__ __launch_bounds__(256)
void gru_l1_scan(const __bf16* __restrict__ y0,
                 const float* __restrict__ wih_f, const float* __restrict__ whh_f,
                 const float* __restrict__ bih_f, const float* __restrict__ bhh_f,
                 const float* __restrict__ wih_b, const float* __restrict__ whh_b,
                 const float* __restrict__ bih_b, const float* __restrict__ bhh_b,
                 float* __restrict__ hfin)
{
    extern __shared__ char smem[];
    __bf16* wxL  = (__bf16*)smem;                    // [384][256] bf16 = 192 KB
    float*  bihL = (float*)(smem + (size_t)G3 * 256 * 2);
    float*  bhhL = bihL + G3;
    float*  gh   = bhhL + G3;                        // [16][384] hidden-proj
    float*  gx   = gh + BT * G3;                     // [16][384] x-proj
    float*  hF   = gx + BT * G3;                     // [16][128]
    __bf16* hB   = (__bf16*)(hF + BT * HH);          // [16][128]
    __bf16* ybuf = hB + BT * HH;                     // [2][16][256] async stage

    const int tid   = threadIdx.x;
    const int lane  = tid & 31;
    const int wave  = tid >> 5;
    const int dir   = blockIdx.x & 1;
    const int bbase = (blockIdx.x >> 1) * BT;

    const float* wih = dir ? wih_b : wih_f;
    const float* whh = dir ? whh_b : whh_f;
    const float* bih = dir ? bih_b : bih_f;
    const float* bhh = dir ? bhh_b : bhh_f;

    for (int i = tid; i < G3 * 256; i += 256) wxL[i] = (__bf16)wih[i];
    for (int i = tid; i < G3; i += 256) { bihL[i] = bih[i]; bhhL[i] = bhh[i]; }
    for (int i = tid; i < BT * HH; i += 256) { hF[i] = 0.f; hB[i] = (__bf16)0.f; }

    const int n_lane = lane & 15;
    const int kB = (lane >> 4) << 4;
    const int kA = (lane >> 4) << 4;

    v16bf Bw[3][4];
#pragma unroll
    for (int i = 0; i < 3; ++i) {
        const int N = wave * NCOL + i * 16 + n_lane;
#pragma unroll
        for (int kk = 0; kk < 4; ++kk) {
            const float* src = whh + (size_t)N * HH + kk * 32 + kB;
            v16bf t;
#pragma unroll
            for (int e = 0; e < 16; ++e) t[e] = (__bf16)src[e];
            Bw[i][kk] = t;
        }
    }

#if HAVE_ASYNC
    {   // pre-stage step 0 rows into ybuf[0]; each wave owns 2 rows
        const int t0 = dir ? (TT - 1) : 0;
        const __bf16* srcbase = y0 + ((size_t)t0 * BB + bbase) * 256;
        const int r0 = wave * 2;
        async_row_copy16(srcbase + (size_t)(r0    ) * 256, ybuf + (r0    ) * 256, lane);
        async_row_copy16(srcbase + (size_t)(r0 + 1) * 256, ybuf + (r0 + 1) * 256, lane);
    }
#endif
    __syncthreads();

    for (int s = 0; s < TT; ++s) {
        const int tstep = dir ? (TT - 1 - s) : s;
        (void)tstep;

#if HAVE_ASYNC
        // kick off next step's y0 rows (double buffer) before the hidden GEMM
        if (s + 1 < TT) {
            const int t2 = dir ? (TT - 2 - s) : (s + 1);
            __bf16* dst = ybuf + ((s + 1) & 1) * (BT * 256);
            const __bf16* srcbase = y0 + ((size_t)t2 * BB + bbase) * 256;
            const int r0 = wave * 2;
            async_row_copy16(srcbase + (size_t)(r0    ) * 256, dst + (r0    ) * 256, lane);
            async_row_copy16(srcbase + (size_t)(r0 + 1) * 256, dst + (r0 + 1) * 256, lane);
        }
#else
        if (s + 1 < TT) {
            const int t2 = dir ? (TT - 2 - s) : (s + 1);
            const __bf16* p = y0 + ((size_t)t2 * BB + (bbase + n_lane)) * 256;
            __builtin_prefetch((const void*)p, 0, 1);
        }
#endif

        // --- hidden projection: K = 128, W_hh in registers ---
        v16bf A[4];
#pragma unroll
        for (int kk = 0; kk < 4; ++kk) {
            const __bf16* hr = hB + n_lane * HH + kk * 32 + kA;
            A[kk] = joinbf(*(const v8bf*)hr, *(const v8bf*)(hr + 8));
        }
        v8f acc[3];
#pragma unroll
        for (int i = 0; i < 3; ++i) acc[i] = vzero8();
#pragma unroll
        for (int kk = 0; kk < 4; ++kk)
#pragma unroll
            for (int i = 0; i < 3; ++i)
                acc[i] = wmma_bf16(A[kk], Bw[i][kk], acc[i]);

        const int mrow = (lane >> 4) << 3;
#pragma unroll
        for (int i = 0; i < 3; ++i) {
            const int N = wave * NCOL + i * 16 + n_lane;
#pragma unroll
            for (int v = 0; v < 8; ++v) gh[(mrow + v) * G3 + N] = acc[i][v];
        }

#if HAVE_ASYNC
        // current buffer's async loads done (only the 2 newest may remain),
        // then make all 16 rows visible to every wave
        if (s + 1 < TT) { WAIT_ASYNC(2); } else { WAIT_ASYNC(0); }
        __syncthreads();
#endif

        // --- input projection: K = 256, A from LDS stage (or global), B from LDS ---
#pragma unroll
        for (int i = 0; i < 3; ++i) acc[i] = vzero8();
#pragma unroll
        for (int kk = 0; kk < 8; ++kk) {
#if HAVE_ASYNC
            const __bf16* yr = ybuf + (s & 1) * (BT * 256) + n_lane * 256 + kk * 32 + kA;
#else
            const __bf16* yr = y0 + ((size_t)tstep * BB + (bbase + n_lane)) * 256 + kk * 32 + kA;
#endif
            const v16bf Ax = joinbf(*(const v8bf*)yr, *(const v8bf*)(yr + 8));
#pragma unroll
            for (int i = 0; i < 3; ++i) {
                const int N = wave * NCOL + i * 16 + n_lane;
                const __bf16* bp = wxL + (size_t)N * 256 + kk * 32 + kB;
                const v16bf Bx = joinbf(*(const v8bf*)bp, *(const v8bf*)(bp + 8));
                acc[i] = wmma_bf16(Ax, Bx, acc[i]);
            }
        }
#pragma unroll
        for (int i = 0; i < 3; ++i) {
            const int N = wave * NCOL + i * 16 + n_lane;
#pragma unroll
            for (int v = 0; v < 8; ++v) gx[(mrow + v) * G3 + N] = acc[i][v];
        }
        __syncthreads();

        // gates: n needs hidden-only term separate from x term
        for (int e = tid; e < BT * HH; e += 256) {
            const int m = e >> 7, j = e & 127;
            const float r  = sigm(gh[m * G3 + j] + bhhL[j] + gx[m * G3 + j] + bihL[j]);
            const float z  = sigm(gh[m * G3 + 128 + j] + bhhL[128 + j] +
                                  gx[m * G3 + 128 + j] + bihL[128 + j]);
            const float hn = gh[m * G3 + 256 + j] + bhhL[256 + j];
            const float xn = gx[m * G3 + 256 + j] + bihL[256 + j];
            const float n  = tanh_fast(xn + r * hn);
            const float hnew = (1.f - z) * n + z * hF[e];
            hF[e] = hnew;
            hB[e] = (__bf16)hnew;
        }
        __syncthreads();
    }

    for (int e = tid; e < BT * HH; e += 256) {
        const int m = e >> 7, j = e & 127;
        hfin[(size_t)(bbase + m) * 256 + dir * HH + j] = hF[e];
    }
}

// ---------------------------------------------------------------------------
// Head: relu(h @ fc1.T + b1) @ fc2.T + b2   (tiny; plain VALU)
// ---------------------------------------------------------------------------
extern "C" __global__ __launch_bounds__(128)
void head_fc(const float* __restrict__ hfin,
             const float* __restrict__ w1, const float* __restrict__ b1,
             const float* __restrict__ w2, const float* __restrict__ b2,
             float* __restrict__ out)
{
    __shared__ float hrow[256];
    __shared__ float a1[128];
    const int b = blockIdx.x, tid = threadIdx.x;
    hrow[tid]       = hfin[(size_t)b * 256 + tid];
    hrow[128 + tid] = hfin[(size_t)b * 256 + 128 + tid];
    __syncthreads();
    float s = b1[tid];
    const float* wr = w1 + (size_t)tid * 256;
#pragma unroll 8
    for (int k = 0; k < 256; ++k) s += wr[k] * hrow[k];
    a1[tid] = fmaxf(s, 0.f);
    __syncthreads();
    if (tid < 2) {
        float o = b2[tid];
        const float* w = w2 + tid * 128;
#pragma unroll 8
        for (int k = 0; k < 128; ++k) o += w[k] * a1[k];
        out[b * 2 + tid] = o;
    }
}

extern "C" void kernel_launch(void* const* d_in, const int* in_sizes, int n_in,
                              void* d_out, int out_size, void* d_ws, size_t ws_size,
                              hipStream_t stream)
{
    (void)in_sizes; (void)n_in; (void)out_size; (void)ws_size;

    const float* x     = (const float*)d_in[0];
    const float* wih0f = (const float*)d_in[1];
    const float* whh0f = (const float*)d_in[2];
    const float* bih0f = (const float*)d_in[3];
    const float* bhh0f = (const float*)d_in[4];
    const float* wih0b = (const float*)d_in[5];
    const float* whh0b = (const float*)d_in[6];
    const float* bih0b = (const float*)d_in[7];
    const float* bhh0b = (const float*)d_in[8];
    const float* wih1f = (const float*)d_in[9];
    const float* whh1f = (const float*)d_in[10];
    const float* bih1f = (const float*)d_in[11];
    const float* bhh1f = (const float*)d_in[12];
    const float* wih1b = (const float*)d_in[13];
    const float* whh1b = (const float*)d_in[14];
    const float* bih1b = (const float*)d_in[15];
    const float* bhh1b = (const float*)d_in[16];
    const float* fc1w  = (const float*)d_in[17];
    const float* fc1b  = (const float*)d_in[18];
    const float* fc2w  = (const float*)d_in[19];
    const float* fc2b  = (const float*)d_in[20];

    // workspace: y0 (bf16 [T][B][256]) then hfin (f32 [B][256])
    __bf16* y0   = (__bf16*)d_ws;
    float*  hfin = (float*)((char*)d_ws + (size_t)TT * BB * 256 * sizeof(__bf16));

    const size_t sm1 = (size_t)(G3 * 4 + G3 + G3 + BT * 4 + BT * G3 + BT * HH) * sizeof(float)
                     + (size_t)BT * HH * sizeof(__bf16);                       // ~46 KB
    const size_t sm2 = (size_t)G3 * 256 * sizeof(__bf16)
                     + (size_t)(G3 + G3 + 2 * BT * G3 + BT * HH) * sizeof(float)
                     + (size_t)BT * HH * sizeof(__bf16)
                     + (size_t)2 * BT * 256 * sizeof(__bf16);                  // ~271 KB (CDNA5 LDS)

    gru_l0_scan<<<dim3(64), dim3(256), sm1, stream>>>(
        x, wih0f, whh0f, bih0f, bhh0f, wih0b, whh0b, bih0b, bhh0b, y0);

    gru_l1_scan<<<dim3(64), dim3(256), sm2, stream>>>(
        y0, wih1f, whh1f, bih1f, bhh1f, wih1b, whh1b, bih1b, bhh1b, hfin);

    head_fc<<<dim3(BB), dim3(128), 0, stream>>>(hfin, fc1w, fc1b, fc2w, fc2b, (float*)d_out);
}
